// MultiHeadAttention_73117523247376
// MI455X (gfx1250) — matrix operand order repaired and checked by reading, not compiled, not run on hardware
//
#include <hip/hip_runtime.h>
#include <hip/hip_bf16.h>

typedef __attribute__((ext_vector_type(16))) _Float16 v16h;
typedef __attribute__((ext_vector_type(8)))  _Float16 v8h;
typedef __attribute__((ext_vector_type(8)))  float    v8f;

#define DIMN 1024
#define SEQL 1024
#define NHEAD 16
#define HDIM 64
#define BATCH 8

// ---- 16-lane xor butterfly via DPP16 ROW_XMASK (0x160|mask): pure VALU ----
template <int MASK>
__device__ __forceinline__ float xor16(float x) {
    return __int_as_float(__builtin_amdgcn_update_dpp(
        0, __float_as_int(x), 0x160 | MASK, 0xF, 0xF, true));
}
__device__ __forceinline__ float wred_max16(float t) {
    t = fmaxf(t, xor16<1>(t));
    t = fmaxf(t, xor16<2>(t));
    t = fmaxf(t, xor16<4>(t));
    t = fmaxf(t, xor16<8>(t));
    return t;
}
__device__ __forceinline__ float wred_sum16(float t) {
    t += xor16<1>(t);
    t += xor16<2>(t);
    t += xor16<4>(t);
    t += xor16<8>(t);
    return t;
}

// ---------------------------------------------------------------------------
// Kernel 1: flash attention. 1 wave = one 16-row Q tile of one (b,h).
// Block = 256 threads (8 waves) -> 128 query rows. Grid = B*H*(L/128) = 1024.
// ---------------------------------------------------------------------------
__global__ __launch_bounds__(256) void attn_flash(
    const float* __restrict__ q, const float* __restrict__ k,
    const float* __restrict__ v, const float* __restrict__ mask,
    _Float16* __restrict__ ctx)
{
    __shared__ _Float16 lds[8][16][40];   // per-wave 16x32 P staging, padded rows

    const int lane  = threadIdx.x & 31;
    const int wave  = threadIdx.x >> 5;
    const int hlf   = lane >> 4;          // which 16-lane half of the wave
    const int ln16  = lane & 15;

    const int blk  = blockIdx.x;
    const int b    = blk >> 7;            // / (H * L/128)
    const int hd   = (blk >> 3) & 15;
    const int tile = blk & 7;
    const int qbase = tile * 128 + wave * 16;

    const float* qrow = q + ((size_t)(b * SEQL + qbase + ln16)) * DIMN + hd * HDIM;
    const float* kbas = k + ((size_t)(b * SEQL)) * DIMN + hd * HDIM;
    const float* vbas = v + ((size_t)(b * SEQL)) * DIMN + hd * HDIM;
    const float* mrow = mask + (size_t)b * SEQL;

    // ---- Q A-fragments (16x32 each, covering d=0..31 and d=32..63) ----
    v16h aq0, aq1;
#pragma unroll
    for (int p = 0; p < 8; ++p) {
        const int kk = p * 2 + 16 * hlf;
        float2 f2 = *(const float2*)(qrow + kk);
        aq0[2 * p] = (_Float16)f2.x; aq0[2 * p + 1] = (_Float16)f2.y;
        f2 = *(const float2*)(qrow + 32 + kk);
        aq1[2 * p] = (_Float16)f2.x; aq1[2 * p + 1] = (_Float16)f2.y;
    }

    float mi[8], li[8];
#pragma unroll
    for (int r = 0; r < 8; ++r) { mi[r] = -3.0e38f; li[r] = 0.0f; }
    v8f acc[4] = {};

    const float inv_scale = 0.125f;  // 1/sqrt(64)

    for (int j = 0; j < SEQL; j += 32) {
        // prefetch next chunk's K/V rows into cache (global_prefetch_b8)
        if (j + 32 < SEQL) {
            __builtin_prefetch(kbas + (size_t)(j + 32 + lane) * DIMN, 0, 3);
            __builtin_prefetch(vbas + (size_t)(j + 32 + lane) * DIMN, 0, 3);
        }

        // ---- scores: two 16x16 tiles (kv cols j..j+15, j+16..j+31) ----
        v8f s[2];
#pragma unroll
        for (int t = 0; t < 2; ++t) {
            const float* krow = kbas + (size_t)(j + t * 16 + ln16) * DIMN;
            v16h b0, b1;
#pragma unroll
            for (int q4 = 0; q4 < 4; ++q4) {
                float4 f0 = *(const float4*)(krow + 16 * hlf + 4 * q4);
                float4 f1 = *(const float4*)(krow + 32 + 16 * hlf + 4 * q4);
                b0[4 * q4 + 0] = (_Float16)f0.x; b0[4 * q4 + 1] = (_Float16)f0.y;
                b0[4 * q4 + 2] = (_Float16)f0.z; b0[4 * q4 + 3] = (_Float16)f0.w;
                b1[4 * q4 + 0] = (_Float16)f1.x; b1[4 * q4 + 1] = (_Float16)f1.y;
                b1[4 * q4 + 2] = (_Float16)f1.z; b1[4 * q4 + 3] = (_Float16)f1.w;
            }
            v8f c = {};
            c = __builtin_amdgcn_wmma_f32_16x16x32_f16(false, aq0, false, b0,
                                                       (short)0, c, false, false);
            c = __builtin_amdgcn_wmma_f32_16x16x32_f16(false, aq1, false, b1,
                                                       (short)0, c, false, false);
            s[t] = c;
        }

        const float mk0 = mrow[j + ln16];
        const float mk1 = mrow[j + 16 + ln16];

        // ---- online softmax, row stats per accumulator register ----
#pragma unroll
        for (int r = 0; r < 8; ++r) {
            float x0 = s[0][r] * inv_scale + mk0;
            float x1 = s[1][r] * inv_scale + mk1;
            float rmax = wred_max16(fmaxf(x0, x1));
            float mnew = fmaxf(mi[r], rmax);
            float corr = __expf(mi[r] - mnew);
            float p0 = __expf(x0 - mnew);
            float p1 = __expf(x1 - mnew);
            float rsum = wred_sum16(p0 + p1);
            li[r] = li[r] * corr + rsum;
            mi[r] = mnew;
            acc[0][r] *= corr; acc[1][r] *= corr;
            acc[2][r] *= corr; acc[3][r] *= corr;
            lds[wave][r + 8 * hlf][ln16]      = (_Float16)p0;
            lds[wave][r + 8 * hlf][16 + ln16] = (_Float16)p1;
        }
        asm volatile("s_wait_dscnt 0" ::: "memory");

        // ---- reload P as A-fragment (16x32) ----
        v16h pA;
        {
            const _Float16* prow = &lds[wave][ln16][0];
            v8h lo = *(const v8h*)(prow + 16 * hlf);
            v8h hi = *(const v8h*)(prow + 16 * hlf + 8);
#pragma unroll
            for (int i = 0; i < 8; ++i) { pA[i] = lo[i]; pA[8 + i] = hi[i]; }
        }

        // ---- O += P * V over four 16-wide d tiles ----
#pragma unroll
        for (int nt = 0; nt < 4; ++nt) {
            const float* vcol = vbas + (size_t)(j + 16 * hlf) * DIMN + nt * 16 + ln16;
            v16h bV;
#pragma unroll
            for (int i = 0; i < 16; ++i) bV[i] = (_Float16)vcol[(size_t)i * DIMN];
            acc[nt] = __builtin_amdgcn_wmma_f32_16x16x32_f16(false, pA, false, bV,
                                                             (short)0, acc[nt], false, false);
        }
    }

    // ---- normalize and emit context as f16 ----
#pragma unroll
    for (int r = 0; r < 8; ++r) {
        const float inv = 1.0f / li[r];
        _Float16* orow = ctx + (size_t)(b * SEQL + qbase + r + 8 * hlf) * DIMN
                             + hd * HDIM + ln16;
        orow[0]  = (_Float16)(acc[0][r] * inv);
        orow[16] = (_Float16)(acc[1][r] * inv);
        orow[32] = (_Float16)(acc[2][r] * inv);
        orow[48] = (_Float16)(acc[3][r] * inv);
    }
}

// ---------------------------------------------------------------------------
// Kernel 2: W_out fp32 -> f16
// ---------------------------------------------------------------------------
__global__ __launch_bounds__(256) void wcvt(const float* __restrict__ w,
                                            _Float16* __restrict__ wh)
{
    const size_t i = (size_t)blockIdx.x * 256 + threadIdx.x;
    wh[i] = (_Float16)w[i];
}

// ---------------------------------------------------------------------------
// Kernel 3: Y = ctx @ W_out^T + b. M=8192, N=1024, K=1024.
// Wave computes 16x64; block (8 waves) 128x64. Grid (64, 16).
// ---------------------------------------------------------------------------
__global__ __launch_bounds__(256) void proj(const _Float16* __restrict__ ctx,
                                            const _Float16* __restrict__ wh,
                                            const float* __restrict__ bias,
                                            float* __restrict__ y)
{
    const int lane = threadIdx.x & 31;
    const int wave = threadIdx.x >> 5;
    const int hlf  = lane >> 4;
    const int ln16 = lane & 15;

    const int rb = blockIdx.x * 128 + wave * 16;
    const int cb = blockIdx.y * 64;

    v8f acc[4] = {};
    for (int kk = 0; kk < DIMN; kk += 32) {
        const _Float16* arow = ctx + (size_t)(rb + ln16) * DIMN + kk;
        v8h lo = *(const v8h*)(arow + 16 * hlf);
        v8h hi = *(const v8h*)(arow + 16 * hlf + 8);
        v16h a;
#pragma unroll
        for (int i = 0; i < 8; ++i) { a[i] = lo[i]; a[8 + i] = hi[i]; }
#pragma unroll
        for (int nt = 0; nt < 4; ++nt) {
            const int col = cb + nt * 16 + ln16;
            const v16h bb = *(const v16h*)(wh + (size_t)col * DIMN + kk + 16 * hlf);
            acc[nt] = __builtin_amdgcn_wmma_f32_16x16x32_f16(false, a, false, bb,
                                                             (short)0, acc[nt], false, false);
        }
    }
#pragma unroll
    for (int nt = 0; nt < 4; ++nt) {
        const int col = cb + nt * 16 + ln16;
        const float bv = bias[col];
#pragma unroll
        for (int r = 0; r < 8; ++r)
            y[(size_t)(rb + r + 8 * hlf) * DIMN + col] = acc[nt][r] + bv;
    }
}

// ---------------------------------------------------------------------------
// Kernel 4: LayerNorm + ELU, one block (256 threads) per row of 1024.
// ---------------------------------------------------------------------------
__global__ __launch_bounds__(256) void ln_elu(const float* __restrict__ y,
                                              const float* __restrict__ gamma,
                                              const float* __restrict__ beta,
                                              float* __restrict__ out)
{
    __shared__ float red0[8], red1[8];
    const int row = blockIdx.x;
    const float* x = y + (size_t)row * DIMN;

    float vals[4];
    float s = 0.0f, s2 = 0.0f;
#pragma unroll
    for (int i = 0; i < 4; ++i) {
        const float t = x[threadIdx.x + i * 256];
        vals[i] = t; s += t; s2 += t * t;
    }
    // 16-lane DPP butterfly, then cross-half via shfl
    s  = wred_sum16(s);
    s2 = wred_sum16(s2);
    s  += __shfl_xor(s, 16, 32);
    s2 += __shfl_xor(s2, 16, 32);

    const int wave = threadIdx.x >> 5, lane = threadIdx.x & 31;
    if (lane == 0) { red0[wave] = s; red1[wave] = s2; }
    __syncthreads();
    s = 0.0f; s2 = 0.0f;
#pragma unroll
    for (int w = 0; w < 8; ++w) { s += red0[w]; s2 += red1[w]; }

    const float mu   = s * (1.0f / DIMN);
    const float var  = s2 * (1.0f / DIMN) - mu * mu;
    const float rstd = rsqrtf(var + 1e-5f);
#pragma unroll
    for (int i = 0; i < 4; ++i) {
        const int c = threadIdx.x + i * 256;
        const float t = (vals[i] - mu) * rstd * gamma[c] + beta[c];
        out[(size_t)row * DIMN + c] = t > 0.0f ? t : (__expf(t) - 1.0f);
    }
}

// ---------------------------------------------------------------------------
extern "C" void kernel_launch(void* const* d_in, const int* in_sizes, int n_in,
                              void* d_out, int out_size, void* d_ws, size_t ws_size,
                              hipStream_t stream)
{
    const float* q     = (const float*)d_in[0];
    const float* k     = (const float*)d_in[1];
    const float* v     = (const float*)d_in[2];
    const float* mask  = (const float*)d_in[3];
    const float* W_out = (const float*)d_in[4];
    const float* b_out = (const float*)d_in[5];
    const float* ln_g  = (const float*)d_in[6];
    const float* ln_b  = (const float*)d_in[7];
    float* out = (float*)d_out;

    char* ws = (char*)d_ws;
    _Float16* ctx = (_Float16*)ws;                                   // 16 MB
    _Float16* wh  = (_Float16*)(ws + (size_t)16 * 1024 * 1024);      //  2 MB
    float*    yb  = (float*)(ws + (size_t)18 * 1024 * 1024);         // 32 MB

    wcvt<<<dim3((DIMN * DIMN) / 256), 256, 0, stream>>>(W_out, wh);
    attn_flash<<<dim3(BATCH * NHEAD * (SEQL / 128)), 256, 0, stream>>>(q, k, v, mask, ctx);
    proj<<<dim3((BATCH * SEQL) / 128, DIMN / 64), 256, 0, stream>>>(ctx, wh, b_out, yb);
    ln_elu<<<dim3(BATCH * SEQL), 256, 0, stream>>>(yb, ln_g, ln_b, out);
}
